// Decoder_66640712565204
// MI455X (gfx1250) — matrix operand order repaired and checked by reading, not compiled, not run on hardware
//
#include <hip/hip_runtime.h>

// ---------------- problem constants (match reference) ----------------
#define BATCH   2
#define LDQ     512
#define LEK     512
#define DMODEL  1024
#define NHEAD   16
#define DKH     64
#define DVH     64
#define DFF     4096
#define NLAYER  4
#define EPS_LN  1e-5f

// ---------------- WMMA types ----------------
typedef _Float16 v16h __attribute__((ext_vector_type(16)));
typedef float    v8f  __attribute__((ext_vector_type(8)));

union Frag16 { v16h v; uint4 u[2]; };

// GEMM tiling: 256 threads = 8 wave32; block tile BM x BN_T, K-step 64.
// CONTRACT: M % 128 == 0, N % BN_T == 0, K % 64 == 0 (true for every launch here).
#define BM   128
#define BK   64
#define LDSS 72   // LDS row stride in halves (BK + 8 pad -> 144B, conflict-free)

struct GemmP {
  const float* A;  int lda; long sAB; long sAH;
  const float* Bm; int ldb; long sBB; long sBH;
  const float* R;                       // residual (same layout as C) or null
  float* C;        int ldc; long sCB; long sCH;
  int M, N, K;
  int batchH;
  float alpha;
  int transB;                           // 1: B operand given as [N x K] row-major
  int relu;
};

// BN_T=128: waves 2(M) x 4(N), wave tile 64x32 = 4x2 WMMA tiles.
// BN_T= 64: waves 4(M) x 2(N), wave tile 32x32 = 2x2 WMMA tiles.
template <int BN_T, int WROWS, int WCOLS>
__global__ __launch_bounds__(256)
void wmma_gemm_kernel(GemmP p) {
  constexpr int TM   = (BM / WROWS) / 16;
  constexpr int TN   = (BN_T / WCOLS) / 16;
  constexpr int AF4  = (BM * BK) / (4 * 256);     // float4 per thread for A stage (8)
  constexpr int BF4  = (BN_T * BK) / (4 * 256);   // float4 per thread for B stage (8 or 4)
  constexpr int NF4K = BN_T / 4;                  // float4 per k-row (non-trans B)

  __shared__ __align__(16) _Float16 As[BM * LDSS];
  __shared__ __align__(16) _Float16 Bs[BN_T * LDSS];

  const int tid   = threadIdx.x;
  const int lane  = tid & 31;
  const int wave  = tid >> 5;
  const int waveM = wave / WCOLS;
  const int waveN = wave % WCOLS;
  const int lgrp  = lane >> 4;          // 0/1 half-wave group
  const int lrow  = lane & 15;

  const int bidx = blockIdx.z / p.batchH;
  const int hidx = blockIdx.z % p.batchH;
  const int blkM = blockIdx.y * BM;
  const int blkN = blockIdx.x * BN_T;

  // Uniform (scalar) base pointers; per-thread offsets are 32-bit & loop-invariant.
  const float* Abase = p.A + (long)bidx * p.sAB + (long)hidx * p.sAH + (long)blkM * p.lda;
  const float* Bbase = p.Bm + (long)bidx * p.sBB + (long)hidx * p.sBH
                     + (p.transB ? (long)blkN * p.ldb : (long)blkN);
  float*       Cg    = p.C + (long)bidx * p.sCB + (long)hidx * p.sCH;
  const float* Rg    = p.R ? (p.R + (long)bidx * p.sCB + (long)hidx * p.sCH) : nullptr;
  const long   bAdv  = p.transB ? 1 : (long)p.ldb;   // global B advance per unit k

  // ---- loop-invariant per-thread staging coordinates ----
  int arow[AF4], ac4[AF4];
#pragma unroll
  for (int i = 0; i < AF4; ++i) {
    int linear = tid + (i << 8);
    arow[i] = linear >> 4;              // 16 float4 per 64-wide row
    ac4[i]  = (linear & 15) << 2;
  }
  int bi0[BF4], bi1[BF4];               // trans: (n, c4)   non-trans: (kr, n4)
#pragma unroll
  for (int i = 0; i < BF4; ++i) {
    int linear = tid + (i << 8);
    if (p.transB) { bi0[i] = linear >> 4;   bi1[i] = (linear & 15) << 2; }
    else          { bi0[i] = linear / NF4K; bi1[i] = (linear % NF4K) << 2; }
  }

  auto loadTile = [&](int k0, float4 (&av)[AF4], float4 (&bv)[BF4]) {
    const float* Ak = Abase + k0;
#pragma unroll
    for (int i = 0; i < AF4; ++i)
      av[i] = *(const float4*)(Ak + arow[i] * p.lda + ac4[i]);
    const float* Bk = Bbase + (long)k0 * bAdv;
    if (p.transB) {
#pragma unroll
      for (int i = 0; i < BF4; ++i)
        bv[i] = *(const float4*)(Bk + bi0[i] * p.ldb + bi1[i]);
    } else {
#pragma unroll
      for (int i = 0; i < BF4; ++i)
        bv[i] = *(const float4*)(Bk + bi0[i] * p.ldb + bi1[i]);
    }
  };

  auto storeTile = [&](const float4 (&av)[AF4], const float4 (&bv)[BF4]) {
#pragma unroll
    for (int i = 0; i < AF4; ++i) {
      _Float16* d = &As[arow[i] * LDSS + ac4[i]];
      d[0] = (_Float16)av[i].x; d[1] = (_Float16)av[i].y;
      d[2] = (_Float16)av[i].z; d[3] = (_Float16)av[i].w;
    }
    if (p.transB) {
#pragma unroll
      for (int i = 0; i < BF4; ++i) {
        _Float16* d = &Bs[bi0[i] * LDSS + bi1[i]];
        d[0] = (_Float16)bv[i].x; d[1] = (_Float16)bv[i].y;
        d[2] = (_Float16)bv[i].z; d[3] = (_Float16)bv[i].w;
      }
    } else {
#pragma unroll
      for (int i = 0; i < BF4; ++i) {
        Bs[(bi1[i] + 0) * LDSS + bi0[i]] = (_Float16)bv[i].x;
        Bs[(bi1[i] + 1) * LDSS + bi0[i]] = (_Float16)bv[i].y;
        Bs[(bi1[i] + 2) * LDSS + bi0[i]] = (_Float16)bv[i].z;
        Bs[(bi1[i] + 3) * LDSS + bi0[i]] = (_Float16)bv[i].w;
      }
    }
  };

  v8f acc[TM][TN];
#pragma unroll
  for (int i = 0; i < TM; ++i)
#pragma unroll
    for (int j = 0; j < TN; ++j) acc[i][j] = {};

  // ---- prologue: stage tile k0=0 ----
  {
    float4 av[AF4], bv[BF4];
    loadTile(0, av, bv);
    storeTile(av, bv);
  }
  __syncthreads();

  for (int k0 = 0; k0 < p.K; k0 += BK) {
    const bool last = (k0 + BK >= p.K);

    // ---- register double-buffer: issue next tile's global loads now ----
    float4 av[AF4], bv[BF4];
    if (!last) loadTile(k0 + BK, av, bv);

    // ---- two K=32 WMMA sub-steps on the staged LDS tile ----
#pragma unroll
    for (int s = 0; s < 2; ++s) {
      Frag16 fa[TM];
#pragma unroll
      for (int tm = 0; tm < TM; ++tm) {
        int m = waveM * (BM / WROWS) + tm * 16 + lrow;
        const _Float16* pa = &As[m * LDSS + 32 * s];
        // A 16x32 f16 layout: lane group g holds K = 8g..8g+7 and 16+8g..+7
        fa[tm].u[0] = *(const uint4*)(pa + lgrp * 8);
        fa[tm].u[1] = *(const uint4*)(pa + 16 + lgrp * 8);
      }
      Frag16 fb[TN];
#pragma unroll
      for (int tn = 0; tn < TN; ++tn) {
        int n = waveN * (BN_T / WCOLS) + tn * 16 + lrow;
        const _Float16* pb = &Bs[n * LDSS + 32 * s + lgrp * 8];
        // B 32x16 f16 layout: lane group g holds K = 8g..8g+7 and 16+8g..+7
        fb[tn].u[0] = *(const uint4*)(pb);
        fb[tn].u[1] = *(const uint4*)(pb + 16);
      }
#pragma unroll
      for (int tm = 0; tm < TM; ++tm)
#pragma unroll
        for (int tn = 0; tn < TN; ++tn)
          acc[tm][tn] = __builtin_amdgcn_wmma_f32_16x16x32_f16(
              false, fa[tm].v, false, fb[tn].v, (short)0, acc[tm][tn], false, false);
    }
    __syncthreads();               // all waves done reading LDS tile

    if (!last) {
      storeTile(av, bv);           // convert f32->f16 into LDS for next round
      __syncthreads();
    }
  }

  // ---- epilogue: alpha, +residual, ReLU (tile-aligned, no guards) ----
#pragma unroll
  for (int tm = 0; tm < TM; ++tm) {
#pragma unroll
    for (int tn = 0; tn < TN; ++tn) {
      int col  = blkN + waveN * (BN_T / WCOLS) + tn * 16 + lrow;
      int rowB = blkM + waveM * (BM / WROWS) + tm * 16 + 8 * lgrp;
#pragma unroll
      for (int vi = 0; vi < 8; ++vi) {
        long idx = (long)(rowB + vi) * p.ldc + col;
        float val = acc[tm][tn][vi] * p.alpha;
        if (Rg) val += Rg[idx];
        if (p.relu) val = fmaxf(val, 0.f);
        Cg[idx] = val;
      }
    }
  }
}

// ---------------- embedding + sinusoidal positional encoding ----------------
__global__ __launch_bounds__(256)
void embed_pe_kernel(const int* __restrict__ toks, const float* __restrict__ emb,
                     float* __restrict__ x, int L, int Dm) {
  const int rowIdx = blockIdx.x;             // b*L + l
  const int l = rowIdx % L;
  const int tok = toks[rowIdx];
  const float* e = emb + (long)tok * Dm;
  float* o = x + (long)rowIdx * Dm;
  const float c = -logf(10000.f) / (float)Dm;
  for (int i = 0; i < Dm / 256; ++i) {
    int d = threadIdx.x + (i << 8);
    int d2 = d & ~1;
    float div = expf((float)d2 * c);
    float ang = (float)l * div;
    float pe = (d & 1) ? cosf(ang) : sinf(ang);
    o[d] = e[d] + pe;
  }
}

// ---------------- masked softmax over last dim (in-place) ----------------
__global__ __launch_bounds__(256)
void softmax_mask_kernel(float* __restrict__ scores, const int* __restrict__ toks,
                         int LQ, int LK, int H, int causal) {
  const int q = blockIdx.x, h = blockIdx.y, b = blockIdx.z;
  float* row = scores + (((long)b * H + h) * LQ + q) * (long)LK;
  const int tid = threadIdx.x;
  __shared__ float red[256];
  float sv[4];
  const int nIt = (LK + 255) >> 8;

  float mx = -3.4e38f;
  for (int i = 0; i < nIt; ++i) {
    int k = tid + (i << 8);
    float s = -1.0e9f;
    if (k < LK) {
      s = row[k];
      bool blocked = (toks[(long)b * LK + k] == 0) || (causal && (k > q));
      if (blocked) s = -1.0e9f;
    }
    sv[i] = s;
    mx = fmaxf(mx, s);
  }
  red[tid] = mx; __syncthreads();
  for (int off = 128; off > 0; off >>= 1) {
    if (tid < off) red[tid] = fmaxf(red[tid], red[tid + off]);
    __syncthreads();
  }
  mx = red[0]; __syncthreads();

  float sum = 0.f;
  for (int i = 0; i < nIt; ++i) {
    int k = tid + (i << 8);
    if (k < LK) { sv[i] = expf(sv[i] - mx); sum += sv[i]; }
  }
  red[tid] = sum; __syncthreads();
  for (int off = 128; off > 0; off >>= 1) {
    if (tid < off) red[tid] += red[tid + off];
    __syncthreads();
  }
  float inv = 1.0f / red[0];
  for (int i = 0; i < nIt; ++i) {
    int k = tid + (i << 8);
    if (k < LK) row[k] = sv[i] * inv;
  }
}

// ---------------- LayerNorm (gamma=1, beta=0, population var) ----------------
__global__ __launch_bounds__(256)
void layernorm_kernel(const float* __restrict__ in, float* __restrict__ out, int Dm) {
  const long row = blockIdx.x;
  const float* p = in + row * Dm;
  float* o = out + row * Dm;
  const int tid = threadIdx.x;
  __shared__ float rs[256], rq[256];
  float s = 0.f, q2 = 0.f;
  for (int i = 0; i < Dm / 256; ++i) {
    float v = p[tid + (i << 8)];
    s += v; q2 += v * v;
  }
  rs[tid] = s; rq[tid] = q2; __syncthreads();
  for (int off = 128; off > 0; off >>= 1) {
    if (tid < off) { rs[tid] += rs[tid + off]; rq[tid] += rq[tid + off]; }
    __syncthreads();
  }
  float mu  = rs[0] / (float)Dm;
  float var = rq[0] / (float)Dm - mu * mu;
  float inv = rsqrtf(var + EPS_LN);
  for (int i = 0; i < Dm / 256; ++i) {
    int d = tid + (i << 8);
    o[d] = (p[d] - mu) * inv;
  }
}

// ---------------- host orchestration ----------------
static inline void launch_gemm(hipStream_t s,
    const float* A, int lda, long sAB, long sAH,
    const float* Bm, int ldb, long sBB, long sBH,
    const float* R, float* C, int ldc, long sCB, long sCH,
    int M, int N, int K, int bB, int bH, float alpha, int transB, int relu) {
  GemmP p{A, lda, sAB, sAH, Bm, ldb, sBB, sBH, R, C, ldc, sCB, sCH,
          M, N, K, bH, alpha, transB, relu};
  if (N <= 64) {
    dim3 g(N / 64, M / BM, bB * bH);
    wmma_gemm_kernel<64, 4, 2><<<g, dim3(256), 0, s>>>(p);
  } else {
    dim3 g(N / 128, M / BM, bB * bH);
    wmma_gemm_kernel<128, 2, 4><<<g, dim3(256), 0, s>>>(p);
  }
}

extern "C" void kernel_launch(void* const* d_in, const int* in_sizes, int n_in,
                              void* d_out, int out_size, void* d_ws, size_t ws_size,
                              hipStream_t stream) {
  (void)in_sizes; (void)n_in; (void)out_size; (void)ws_size;

  const int*   dec    = (const int*)d_in[0];
  const int*   enc    = (const int*)d_in[1];
  const float* encOut = (const float*)d_in[2];
  const float* emb    = (const float*)d_in[3];
  const float* sWq    = (const float*)d_in[4];
  const float* sWk    = (const float*)d_in[5];
  const float* sWv    = (const float*)d_in[6];
  const float* sWo    = (const float*)d_in[7];
  const float* cWq    = (const float*)d_in[8];
  const float* cWk    = (const float*)d_in[9];
  const float* cWv    = (const float*)d_in[10];
  const float* cWo    = (const float*)d_in[11];
  const float* fW1    = (const float*)d_in[12];
  const float* fW2    = (const float*)d_in[13];
  float* out = (float*)d_out;
  float* ws  = (float*)d_ws;

  const long M1 = 1024l * 1024;             // 1M floats
  float* x   = ws;                          // [1024,1024]
  float* y   = ws + 1 * M1;
  float* qb  = ws + 2 * M1;
  float* kb  = ws + 3 * M1;
  float* vb  = ws + 4 * M1;
  float* big = ws + 5 * M1;                 // scores (8M floats) / ffn hidden (4M)
  float* ctx = qb;                          // reuse q after scores are formed

  const int ROWS = BATCH * LDQ;             // 1024
  const long WATT = (long)DMODEL * NHEAD * DKH;   // 1M per layer
  const long WFF  = (long)DMODEL * DFF;           // 4M per layer
  const long sQB  = (long)LDQ * DMODEL;           // per-batch stride in q/k/v/ctx
  const long sSB  = (long)NHEAD * LDQ * LEK;      // per-batch stride in scores
  const long sSH  = (long)LDQ * LEK;              // per-head stride in scores
  const float iscale = 0.125f;                    // 1/sqrt(64)

  // x = emb[dec] + PE
  embed_pe_kernel<<<dim3(ROWS), dim3(256), 0, stream>>>(dec, emb, x, LDQ, DMODEL);

  for (int l = 0; l < NLAYER; ++l) {
    // ---------- self attention ----------
    launch_gemm(stream, x, DMODEL, 0, 0, sWq + l * WATT, DMODEL, 0, 0, nullptr,
                qb, DMODEL, 0, 0, ROWS, DMODEL, DMODEL, 1, 1, 1.f, 0, 0);
    launch_gemm(stream, x, DMODEL, 0, 0, sWk + l * WATT, DMODEL, 0, 0, nullptr,
                kb, DMODEL, 0, 0, ROWS, DMODEL, DMODEL, 1, 1, 1.f, 0, 0);
    launch_gemm(stream, x, DMODEL, 0, 0, sWv + l * WATT, DMODEL, 0, 0, nullptr,
                vb, DMODEL, 0, 0, ROWS, DMODEL, DMODEL, 1, 1, 1.f, 0, 0);
    // scores = (Q @ K^T) / sqrt(dk)   [B,H,LQ,LK]
    launch_gemm(stream, qb, DMODEL, sQB, DKH, kb, DMODEL, sQB, DKH, nullptr,
                big, LEK, sSB, sSH, LDQ, LEK, DKH, BATCH, NHEAD, iscale, 1, 0);
    softmax_mask_kernel<<<dim3(LDQ, NHEAD, BATCH), dim3(256), 0, stream>>>(
        big, dec, LDQ, LEK, NHEAD, /*causal=*/1);
    // ctx = attn @ V  -> written directly as [B, LQ, H*DV]
    launch_gemm(stream, big, LEK, sSB, sSH, vb, DMODEL, sQB, DVH, nullptr,
                ctx, DMODEL, sQB, DVH, LDQ, DVH, LEK, BATCH, NHEAD, 1.f, 0, 0);
    // y = ctx @ Wo + x ; x = LN(y)
    launch_gemm(stream, ctx, DMODEL, 0, 0, sWo + l * WATT, DMODEL, 0, 0, x,
                y, DMODEL, 0, 0, ROWS, DMODEL, DMODEL, 1, 1, 1.f, 0, 0);
    layernorm_kernel<<<dim3(ROWS), dim3(256), 0, stream>>>(y, x, DMODEL);

    // ---------- cross attention ----------
    launch_gemm(stream, x, DMODEL, 0, 0, cWq + l * WATT, DMODEL, 0, 0, nullptr,
                qb, DMODEL, 0, 0, ROWS, DMODEL, DMODEL, 1, 1, 1.f, 0, 0);
    launch_gemm(stream, encOut, DMODEL, 0, 0, cWk + l * WATT, DMODEL, 0, 0, nullptr,
                kb, DMODEL, 0, 0, ROWS, DMODEL, DMODEL, 1, 1, 1.f, 0, 0);
    launch_gemm(stream, encOut, DMODEL, 0, 0, cWv + l * WATT, DMODEL, 0, 0, nullptr,
                vb, DMODEL, 0, 0, ROWS, DMODEL, DMODEL, 1, 1, 1.f, 0, 0);
    launch_gemm(stream, qb, DMODEL, sQB, DKH, kb, DMODEL, sQB, DKH, nullptr,
                big, LEK, sSB, sSH, LDQ, LEK, DKH, BATCH, NHEAD, iscale, 1, 0);
    softmax_mask_kernel<<<dim3(LDQ, NHEAD, BATCH), dim3(256), 0, stream>>>(
        big, enc, LDQ, LEK, NHEAD, /*causal=*/0);
    launch_gemm(stream, big, LEK, sSB, sSH, vb, DMODEL, sQB, DVH, nullptr,
                ctx, DMODEL, sQB, DVH, LDQ, DVH, LEK, BATCH, NHEAD, 1.f, 0, 0);
    launch_gemm(stream, ctx, DMODEL, 0, 0, cWo + l * WATT, DMODEL, 0, 0, x,
                y, DMODEL, 0, 0, ROWS, DMODEL, DMODEL, 1, 1, 1.f, 0, 0);
    layernorm_kernel<<<dim3(ROWS), dim3(256), 0, stream>>>(y, x, DMODEL);

    // ---------- FFN ----------
    launch_gemm(stream, x, DMODEL, 0, 0, fW1 + l * WFF, DFF, 0, 0, nullptr,
                big, DFF, 0, 0, ROWS, DFF, DMODEL, 1, 1, 1.f, 0, /*relu=*/1);
    launch_gemm(stream, big, DFF, 0, 0, fW2 + l * WFF, DMODEL, 0, 0, x,
                y, DMODEL, 0, 0, ROWS, DMODEL, DFF, 1, 1, 1.f, 0, 0);
    layernorm_kernel<<<dim3(ROWS), dim3(256), 0, stream>>>(
        y, (l == NLAYER - 1) ? out : x, DMODEL);
  }
}
